// Model_25400436588948
// MI455X (gfx1250) — matrix operand order repaired and checked by reading, not compiled, not run on hardware
//
#include <hip/hip_runtime.h>

typedef __attribute__((ext_vector_type(16))) _Float16 v16h;
typedef __attribute__((ext_vector_type(8)))  _Float16 v8h;
typedef __attribute__((ext_vector_type(8)))  float    v8f;

#define LVL  10
#define TBLS 65536u
#define ROWS_PB 128
#define NTHREADS 256

// Branchless SELU using the fast exp intrinsic (v_exp_f32, no range-check branches).
__device__ __forceinline__ float selu_f(float x) {
    const float sc = 1.0507009873554805f;
    const float al = 1.6732632423543772f;
    float e = __expf(fminf(x, 0.f));        // clamp avoids overflow; exact for x<=0
    float neg = al * (e - 1.f);
    return sc * (x > 0.f ? x : neg);
}

__device__ __forceinline__ v8f wmma16x16x32(v16h a, v16h b, v8f c) {
    // v_wmma_f32_16x16x32_f16 : D = A(16x32 f16) * B(32x16 f16) + C(16x16 f32)
    return __builtin_amdgcn_wmma_f32_16x16x32_f16(
        /*neg_a=*/false, a, /*neg_b=*/false, b,
        /*c_mod=*/(short)0, c, /*reuse_a=*/false, /*reuse_b=*/false);
}

// A-fragment per ISA layout: lane holds row (wr0 + lane&15), kb=(lane>>4)*8,
// halves 0..7 = K kb..kb+7, halves 8..15 = K kb+16..kb+23 (within a 32-K step).
__device__ __forceinline__ v16h load_a_frag(const _Float16* buf, int row, int koff) {
    const v8h* p = (const v8h*)(buf + row * 64 + koff);
    v8h lo = p[0];
    v8h hi = p[2];   // +16 halves
    return __builtin_shufflevector(lo, hi, 0,1,2,3,4,5,6,7,8,9,10,11,12,13,14,15);
}

__device__ __forceinline__ void mlp_layer(const _Float16* inb, const _Float16* Wsw,
                                          const float* bias, _Float16* outb,
                                          int wr0, int lane) {
    const int mrow   = wr0 + (lane & 15);
    const int kb     = (lane >> 4) * 8;
    const int rshift = (lane >> 4) * 8;
    v16h a0 = load_a_frag(inb, mrow, 0  + kb);
    v16h a1 = load_a_frag(inb, mrow, 32 + kb);
#pragma unroll
    for (int nt = 0; nt < 4; ++nt) {
        v8f c = {};
        v16h b0 = *(const v16h*)(Wsw + (((0 * 4 + nt) * 32) + lane) * 16);
        v16h b1 = *(const v16h*)(Wsw + (((1 * 4 + nt) * 32) + lane) * 16);
        c = wmma16x16x32(a0, b0, c);
        c = wmma16x16x32(a1, b1, c);
        const int col = nt * 16 + (lane & 15);
        const float bs = bias[col];
#pragma unroll
        for (int r = 0; r < 8; ++r) {
            int R = wr0 + r + rshift;
            outb[R * 64 + col] = (_Float16)selu_f(c[r] + bs);
        }
    }
}

__global__ __launch_bounds__(NTHREADS)
void fused_hashmlp_kernel(const float* __restrict__ vertices,
                          const int*   __restrict__ indices,
                          const float* __restrict__ center,
                          const float* __restrict__ scene_scaling,
                          const float* __restrict__ table,
                          const float* __restrict__ W1, const float* __restrict__ b1,
                          const float* __restrict__ W2, const float* __restrict__ b2,
                          const float* __restrict__ W3, const float* __restrict__ b3,
                          float* __restrict__ out, int M) {
    __shared__ __align__(32) _Float16 sA[ROWS_PB * 64];   // features / h2
    __shared__ __align__(32) _Float16 sB[ROWS_PB * 64];   // h1
    __shared__ __align__(32) _Float16 sW1[64 * 64];       // B-frag swizzled
    __shared__ __align__(32) _Float16 sW2[64 * 64];       // B-frag swizzled
    __shared__ __align__(32) _Float16 sW3f[2 * 32 * 16];  // W3 padded to 64x16, B-frag order
    __shared__ float sb1[64], sb2[64];
    __shared__ float sb3;

    const int t = threadIdx.x;

    // ---- Stage weights into LDS, pre-swizzled into WMMA B-fragment order ----
    {
        const int ks = t >> 7, nt = (t >> 5) & 3, lane = t & 31;
        const int kb8 = 8 * (lane >> 4);
        const int N = nt * 16 + (lane & 15);
        const int base = t * 16;
#pragma unroll
        for (int j = 0; j < 16; ++j) {
            int Kg = ks * 32 + kb8 + j + ((j >> 3) << 3);
            float w1v = (Kg < 40) ? W1[Kg * 64 + N] : 0.f;  // pad K 40..63
            sW1[base + j] = (_Float16)w1v;
            sW2[base + j] = (_Float16)W2[Kg * 64 + N];
        }
        // W3 (64x1) -> 64x16 B operand, only column 0 nonzero. 2 ksteps x 32 lanes.
        if (t < 64) {
            const int ks3 = t >> 5, ln3 = t & 31;
            const bool col0 = ((ln3 & 15) == 0);
#pragma unroll
            for (int j = 0; j < 16; ++j) {
                int Kg = ks3 * 32 + 8 * (ln3 >> 4) + j + ((j >> 3) << 3);
                sW3f[(ks3 * 32 + ln3) * 16 + j] = col0 ? (_Float16)W3[Kg] : (_Float16)0.f;
            }
            sb1[t] = b1[t]; sb2[t] = b2[t];
        }
        if (t == 0) sb3 = b3[0];
    }

    // ---- Geometry + hashgrid encode: 2 threads per sample (5 levels each) ----
    {
        const int rowL = t & (ROWS_PB - 1);
        const int half = t >> 7;
        long m = (long)blockIdx.x * ROWS_PB + rowL;
        if (m >= M) m = M - 1;                         // clamp reads only
        const int* ix = indices + m * 4;
        float vx[4], vy[4], vz[4];
#pragma unroll
        for (int i = 0; i < 4; ++i) {
            const float* vp = vertices + (long)ix[i] * 3;
            vx[i] = vp[0]; vy[i] = vp[1]; vz[i] = vp[2];
        }
        // rows of A = v_i - v_0 ; rhs_i = 0.5*|a_i|^2
        float a1x = vx[1]-vx[0], a1y = vy[1]-vy[0], a1z = vz[1]-vz[0];
        float a2x = vx[2]-vx[0], a2y = vy[2]-vy[0], a2z = vz[2]-vz[0];
        float a3x = vx[3]-vx[0], a3y = vy[3]-vy[0], a3z = vz[3]-vz[0];
        float r1 = 0.5f*(a1x*a1x + a1y*a1y + a1z*a1z);
        float r2 = 0.5f*(a2x*a2x + a2y*a2y + a2z*a2z);
        float r3 = 0.5f*(a3x*a3x + a3y*a3y + a3z*a3z);
        // Cramer: x = (b1*(a2xa3) + b2*(a3xa1) + b3*(a1xa2)) / det
        float c23x = a2y*a3z - a2z*a3y, c23y = a2z*a3x - a2x*a3z, c23z = a2x*a3y - a2y*a3x;
        float c31x = a3y*a1z - a3z*a1y, c31y = a3z*a1x - a3x*a1z, c31z = a3x*a1y - a3y*a1x;
        float c12x = a1y*a2z - a1z*a2y, c12y = a1z*a2x - a1x*a2z, c12z = a1x*a2y - a1y*a2x;
        float det = a1x*c23x + a1y*c23y + a1z*c23z;
        float inv = 1.f / det;
        float cx = (r1*c23x + r2*c31x + r3*c12x) * inv;
        float cy = (r1*c23y + r2*c31y + r3*c12y) * inv;
        float cz = (r1*c23z + r2*c31z + r3*c12z) * inv;
        float radius = sqrtf(cx*cx + cy*cy + cz*cz + 1e-12f);
        const float ss = scene_scaling[0];
        float nx = (vx[0] + cx - center[0]) / ss;
        float ny = (vy[0] + cy - center[1]) / ss;
        float nz = (vz[0] + cz - center[2]) / ss;
        float nrm = sqrtf(nx*nx + ny*ny + nz*nz);
        float sn  = fmaxf(nrm, 1.f);
        float cs  = (nrm <= 1.f) ? 1.f : (2.f - 1.f/sn) / sn;  // contraction scale
        float cvx = nx*cs, cvy = ny*cs, cvz = nz*cs;
        float cr  = ((nrm <= 1.f) ? (radius/ss) : (radius/(ss*sn*sn))) * 0.5f;
        float px = fminf(fmaxf((cvx*0.5f + 1.f)*0.5f, 0.f), 1.f - 1e-6f);
        float py = fminf(fmaxf((cvy*0.5f + 1.f)*0.5f, 0.f), 1.f - 1e-6f);
        float pz = fminf(fmaxf((cvz*0.5f + 1.f)*0.5f, 0.f), 1.f - 1e-6f);

        for (int li = 0; li < 5; ++li) {
            const int l = half * 5 + li;
            const float res = 16.f * (float)(1 << l);
            float xs0 = px*res, xs1 = py*res, xs2 = pz*res;
            float f0 = floorf(xs0), f1 = floorf(xs1), f2 = floorf(xs2);
            float w0 = xs0 - f0, w1 = xs1 - f1, w2 = xs2 - f2;
            unsigned xi0 = (unsigned)f0, xi1 = (unsigned)f1, xi2 = (unsigned)f2;
            float d2  = fmaxf(8.f * (float)l * cr, 1e-12f);
            float den = fmaxf(sqrtf(d2), 1e-12f);
            float scl = erff(1.f / den);
            float acc0 = 0.f, acc1 = 0.f, acc2 = 0.f, acc3 = 0.f;
#pragma unroll
            for (int cnr = 0; cnr < 8; ++cnr) {
                unsigned ox = (cnr >> 2) & 1u, oy = (cnr >> 1) & 1u, oz = cnr & 1u;
                unsigned h = (xi0 + ox)
                           ^ ((xi1 + oy) * 2654435761u)
                           ^ ((xi2 + oz) * 805459861u);
                h &= (TBLS - 1u);
                float wt = (ox ? w0 : 1.f - w0) * (oy ? w1 : 1.f - w1) * (oz ? w2 : 1.f - w2);
                const float* e = table + ((unsigned)l * TBLS + h) * 4u;
                acc0 += wt * e[0]; acc1 += wt * e[1];
                acc2 += wt * e[2]; acc3 += wt * e[3];
            }
            // feat index f = d*L + l, scaled per level
            sA[rowL * 64 + 0 * LVL + l] = (_Float16)(acc0 * scl);
            sA[rowL * 64 + 1 * LVL + l] = (_Float16)(acc1 * scl);
            sA[rowL * 64 + 2 * LVL + l] = (_Float16)(acc2 * scl);
            sA[rowL * 64 + 3 * LVL + l] = (_Float16)(acc3 * scl);
        }
        if (half == 1) {
#pragma unroll
            for (int f = 40; f < 64; ++f) sA[rowL * 64 + f] = (_Float16)0.f;
        }
    }
    __syncthreads();

    // ---- MLP layers 1 & 2 via WMMA (each wave owns 16 rows) ----
    const int lane = t & 31;
    const int wr0  = (t >> 5) * 16;
    mlp_layer(sA, sW1, sb1, sB, wr0, lane);   // 40(->64) -> 64, selu
    __syncthreads();
    mlp_layer(sB, sW2, sb2, sA, wr0, lane);   // 64 -> 64, selu
    __syncthreads();

    // ---- Layer 3 via WMMA: [16x64] @ [64x16] with only column 0 of B nonzero ----
    {
        const int mrow = wr0 + (lane & 15);
        const int kb   = (lane >> 4) * 8;
        v16h a0 = load_a_frag(sA, mrow, 0  + kb);
        v16h a1 = load_a_frag(sA, mrow, 32 + kb);
        v8f c = {};
        c = wmma16x16x32(a0, *(const v16h*)(sW3f + (0 * 32 + lane) * 16), c);
        c = wmma16x16x32(a1, *(const v16h*)(sW3f + (1 * 32 + lane) * 16), c);
        // Column 0 lives in lanes with (lane & 15) == 0; vgpr r -> row r + 8*(lane>>4).
        if ((lane & 15) == 0) {
            const float bb = sb3;
            const int rsh = (lane >> 4) * 8;
#pragma unroll
            for (int r = 0; r < 8; ++r) {
                long m = (long)blockIdx.x * ROWS_PB + wr0 + r + rsh;
                if (m < M) out[m] = c[r] + bb;
            }
        }
    }
}

extern "C" void kernel_launch(void* const* d_in, const int* in_sizes, int n_in,
                              void* d_out, int out_size, void* d_ws, size_t ws_size,
                              hipStream_t stream) {
    (void)n_in; (void)d_ws; (void)ws_size; (void)out_size;
    const float* vertices = (const float*)d_in[0];
    const int*   indices  = (const int*)d_in[1];
    const float* center   = (const float*)d_in[2];
    const float* ss       = (const float*)d_in[3];
    const float* table    = (const float*)d_in[4];
    const float* W1       = (const float*)d_in[5];
    const float* b1       = (const float*)d_in[6];
    const float* W2       = (const float*)d_in[7];
    const float* b2       = (const float*)d_in[8];
    const float* W3       = (const float*)d_in[9];
    const float* b3       = (const float*)d_in[10];
    float* out = (float*)d_out;

    const int M = in_sizes[1] / 4;
    const int blocks = (M + ROWS_PB - 1) / ROWS_PB;
    fused_hashmlp_kernel<<<blocks, NTHREADS, 0, stream>>>(
        vertices, indices, center, ss, table, W1, b1, W2, b2, W3, b3, out, M);
}
